// DACAM_4234837754250
// MI455X (gfx1250) — matrix operand order repaired and checked by reading, not compiled, not run on hardware
//
#include <hip/hip_runtime.h>

// ---------------- types / constants ----------------
typedef __attribute__((ext_vector_type(16))) __bf16 v16bf;
typedef __attribute__((ext_vector_type(8)))  __bf16 v8bf;
typedef __attribute__((ext_vector_type(8)))  float  v8f;

#define NB   32          // batches
#define NCH  2048        // channels C
#define HWP  784         // H*W pixels
#define NCL  200         // num classes
#define PT   49          // pixel tiles of 16 (49*16 = 784)
#define NSAMP 32         // SAMPLE_NUM

// workspace layout (bytes)
#define FEATS_BYTES  ((size_t)NB * HWP * NCH * 2)          // bf16 [B][HW][C]
#define S2_BYTES     ((size_t)NB * HWP * 4)                // f32  [B][HW]
#define TOT_BYTES    ((size_t)NB * 64 * NCH * 2)           // bf16 [B][64][C]

__device__ inline v16bf cat8(v8bf a, v8bf b) {
  return __builtin_shufflevector(a, b, 0,1,2,3,4,5,6,7,8,9,10,11,12,13,14,15);
}

// ---------------- kernel A: transpose f32 [C][HW] -> bf16 [HW][C], fused s2 ----------------
__global__ void dacam_transpose_kernel(const float* __restrict__ px,
                                       __bf16* __restrict__ featsT,
                                       float* __restrict__ s2,
                                       float* __restrict__ accums) {
  const int blk = blockIdx.x;           // B * 49 blocks
  const int b  = blk / PT;
  const int p0 = (blk % PT) * 16;
  const int tid = threadIdx.x;
  const int ci = tid >> 4;              // channel-in-tile for loads
  const int pi = tid & 15;              // pixel-in-tile for loads
  __shared__ float tile[16][17];
  __shared__ float s2s[16];
  if (tid < 16) s2s[tid] = 0.0f;
  __syncthreads();

  const float* base = px + (size_t)b * NCH * HWP;
  __bf16* obase = featsT + (size_t)b * HWP * NCH;
  float acc = 0.0f;
  const int pi2 = tid >> 4;             // pixel-in-tile for stores
  const int ci2 = tid & 15;             // channel-in-tile for stores
  for (int c0 = 0; c0 < NCH; c0 += 16) {
    float v = base[(size_t)(c0 + ci) * HWP + p0 + pi];
    acc += v * v;
    tile[ci][pi] = v;
    __syncthreads();
    obase[(size_t)(p0 + pi2) * NCH + c0 + ci2] = (__bf16)tile[ci2][pi2];
    __syncthreads();
  }
  atomicAdd(&s2s[pi], acc);
  __syncthreads();
  if (tid < 16) s2[b * HWP + p0 + tid] = s2s[tid];
  if (blk == 0 && tid < 8) accums[tid] = 0.0f;   // zero loss accumulators
}

// ---------------- kernel B: per-batch k-means + selection + loss_u + gather totals ----------------
__global__ void dacam_kmeans_kernel(const __bf16* __restrict__ featsT,
                                    const float* __restrict__ s2g,
                                    const float* __restrict__ imgf,
                                    const float* __restrict__ ccent,
                                    const float* __restrict__ ucent,
                                    const int*   __restrict__ target,
                                    __bf16* __restrict__ totals,
                                    float* __restrict__ accums) {
  const int b   = blockIdx.x;
  const int tid = threadIdx.x;
  const int wave = tid >> 5;
  const int lane = tid & 31;
  const __bf16* fb = featsT + (size_t)b * HWP * NCH;

  __shared__ float centF[3][NCH];                    // fp32 centers
  __shared__ __align__(32) __bf16 centBp[4][NCH];    // bf16 mirror, row 3 = zeros
  __shared__ float s2s[HWP];
  __shared__ int   lab[HWP];
  __shared__ float wdot[8][64];                      // per-wave dot scratch [m][k]
  __shared__ float c2[3];
  __shared__ int   cntS[3];
  __shared__ int   selS[NSAMP - 1];
  __shared__ int   selT[NSAMP];
  __shared__ float redF;
  __shared__ int   redI;

  // ---- init centers (+ zero pad row) ----
  const int tg = target[b];
  for (int c = tid; c < NCH; c += 256) {
    float v0 = ucent[c];
    float v1 = imgf[(size_t)b * NCH + c];
    float v2 = ccent[(size_t)tg * NCH + c];
    centF[0][c] = v0; centF[1][c] = v1; centF[2][c] = v2;
    centBp[0][c] = (__bf16)v0; centBp[1][c] = (__bf16)v1;
    centBp[2][c] = (__bf16)v2; centBp[3][c] = (__bf16)0.0f;
  }
  for (int p = tid; p < HWP; p += 256) s2s[p] = s2g[b * HWP + p];
  __syncthreads();

  const int mrow = lane & 15;
  const int rsel = (mrow < 3) ? mrow : 3;            // padded B row (3 => zeros)
  const int aO0 = (lane < 16) ? 0 : 16;
  const int aO1 = aO0 + 8;
  const int bO  = (lane < 16) ? 0 : 16;
  const int mBase = (lane < 16) ? 0 : 8;

  for (int it = 0; it <= 10; ++it) {
    // ---- c2[k] = |center_k|^2 ----
    if (tid < 3) c2[tid] = 0.0f;
    __syncthreads();
    for (int k = 0; k < 3; ++k) {
      float part = 0.0f;
      for (int c = tid; c < NCH; c += 256) { float v = centF[k][c]; part += v * v; }
      atomicAdd(&c2[k], part);
    }
    __syncthreads();

    // ---- label pass: paired tiles share each B chunk (2 WMMAs per ds load) ----
    for (int t0 = wave; t0 < 56; t0 += 16) {
      const int t1 = t0 + 8;                          // virtual tiles; >=49 discarded
      const int pr0 = min(t0 * 16 + mrow, HWP - 1);   // row-clamped (stays in-bounds)
      const int pr1 = min(t1 * 16 + mrow, HWP - 1);
      const __bf16* ar0 = fb + (size_t)pr0 * NCH;
      const __bf16* ar1 = fb + (size_t)pr1 * NCH;
      __builtin_prefetch(ar0, 0, 1);
      __builtin_prefetch(ar1, 0, 1);
      v8f acc0 = {};
      v8f acc1 = {};
      for (int k0 = 0; k0 < NCH; k0 += 32) {
        v16bf Bm = *(const v16bf*)(&centBp[rsel][k0 + bO]);   // unconditional
        v16bf A0 = cat8(*(const v8bf*)(ar0 + k0 + aO0),
                        *(const v8bf*)(ar0 + k0 + aO1));
        v16bf A1 = cat8(*(const v8bf*)(ar1 + k0 + aO0),
                        *(const v8bf*)(ar1 + k0 + aO1));
        acc0 = __builtin_amdgcn_wmma_f32_16x16x32_bf16(
            false, A0, false, Bm, (short)0, acc0, false, false);
        acc1 = __builtin_amdgcn_wmma_f32_16x16x32_bf16(
            false, A1, false, Bm, (short)0, acc1, false, false);
      }
      // scatter dots + argmin labels for both tiles
#pragma unroll
      for (int half = 0; half < 2; ++half) {
        const v8f acc = half ? acc1 : acc0;
        const int t = half ? t1 : t0;
        if (mrow < 3) {
#pragma unroll
          for (int r = 0; r < 8; ++r) wdot[wave][(r + mBase) * 4 + mrow] = acc[r];
        }
        __builtin_amdgcn_wave_barrier();
        if (lane < 16) {
          const int p = t * 16 + lane;
          if (p < HWP) {
            const float s2v = s2s[p];
            float d0 = s2v - 2.0f * wdot[wave][lane * 4 + 0] + c2[0];
            float d1 = s2v - 2.0f * wdot[wave][lane * 4 + 1] + c2[1];
            float d2 = s2v - 2.0f * wdot[wave][lane * 4 + 2] + c2[2];
            int bl = 0; float bd = d0;
            if (d1 < bd) { bd = d1; bl = 1; }
            if (d2 < bd) { bd = d2; bl = 2; }
            lab[p] = bl;
          }
        }
        __builtin_amdgcn_wave_barrier();
      }
    }
    __syncthreads();

    if (it == 10) break;   // final labels with post-loop centers, no update

    // ---- center update ----
    if (tid < 3) cntS[tid] = 0;
    __syncthreads();
    for (int p = tid; p < HWP; p += 256) atomicAdd(&cntS[lab[p]], 1);
    __syncthreads();

    float a0s[8], a1s[8], a2s[8];
#pragma unroll
    for (int j = 0; j < 8; ++j) { a0s[j] = 0.f; a1s[j] = 0.f; a2s[j] = 0.f; }
    for (int p = 0; p < HWP; ++p) {
      const int l = lab[p];
      const float m0 = (l == 0) ? 1.f : 0.f;
      const float m1 = (l == 1) ? 1.f : 0.f;
      const float m2 = (l == 2) ? 1.f : 0.f;
      const v8bf v = ((const v8bf*)(fb + (size_t)p * NCH))[tid];
#pragma unroll
      for (int j = 0; j < 8; ++j) {
        float f = (float)v[j];
        a0s[j] += m0 * f; a1s[j] += m1 * f; a2s[j] += m2 * f;
      }
    }
    const int c0 = tid * 8;
    const float n0 = fmaxf((float)cntS[0], 1.f);
    const float n1 = fmaxf((float)cntS[1], 1.f);
    const float n2 = fmaxf((float)cntS[2], 1.f);
#pragma unroll
    for (int j = 0; j < 8; ++j) {
      float v0 = (cntS[0] > 0) ? a0s[j] / n0 : centF[0][c0 + j];
      float v1 = (cntS[1] > 0) ? a1s[j] / n1 : centF[1][c0 + j];
      float v2 = (cntS[2] > 0) ? a2s[j] / n2 : centF[2][c0 + j];
      centF[0][c0 + j] = v0; centBp[0][c0 + j] = (__bf16)v0;
      centF[1][c0 + j] = v1; centBp[1][c0 + j] = (__bf16)v1;
      centF[2][c0 + j] = v2; centBp[2][c0 + j] = (__bf16)v2;
    }
    __syncthreads();
  }

  // ---- deterministic selection (stable-argsort semantics) ----
  if (tid == 0) {
    int ns = 0, nt = 0;
    for (int p = 0; p < HWP; ++p) {
      if (lab[p] == 1 && ns < NSAMP - 1) selS[ns++] = p;
      if (lab[p] == 2 && nt < NSAMP)     selT[nt++] = p;
    }
    for (int p = 0; p < HWP && ns < NSAMP - 1; ++p) if (lab[p] != 1) selS[ns++] = p;
    for (int p = 0; p < HWP && nt < NSAMP;     ++p) if (lab[p] != 2) selT[nt++] = p;
  }
  __syncthreads();

  // ---- gather total = [img_feat ; src(31) ; tgt(32)] as bf16 [64][C] ----
  __bf16* Tb = totals + (size_t)b * 64 * NCH;
  for (int c = tid; c < NCH; c += 256) Tb[c] = (__bf16)imgf[(size_t)b * NCH + c];
  for (int r = 1; r < NSAMP; ++r) {
    const int p = selS[r - 1];
    for (int c = tid; c < NCH; c += 256) Tb[(size_t)r * NCH + c] = fb[(size_t)p * NCH + c];
  }
  for (int r = 0; r < NSAMP; ++r) {
    const int p = selT[r];
    for (int c = tid; c < NCH; c += 256) Tb[(size_t)(NSAMP + r) * NCH + c] = fb[(size_t)p * NCH + c];
  }

  // ---- universum loss partials ----
  float upart = 0.0f; int ucnt = 0;
  for (int p = tid; p < HWP; p += 256) {
    if (lab[p] == 0) {
      ++ucnt;
      const v8bf* rv = (const v8bf*)(fb + (size_t)p * NCH);
      float s = 0.0f;
      for (int q = 0; q < NCH / 8; ++q) {
        v8bf v = rv[q];
#pragma unroll
        for (int j = 0; j < 8; ++j) s += (float)v[j];
      }
      upart += s;
    }
  }
  if (tid == 0) { redF = 0.0f; redI = 0; }
  __syncthreads();
  atomicAdd(&redF, upart);
  atomicAdd(&redI, ucnt);
  __syncthreads();
  if (tid == 0) {
    atomicAdd(&accums[0], redF);
    atomicAdd(&accums[1], (float)redI);
  }
}

// ---------------- kernel C: per-batch RBF-MMD via WMMA Gram matrix ----------------
__global__ void dacam_mmd_kernel(const __bf16* __restrict__ totals,
                                 float* __restrict__ accums) {
  const int b   = blockIdx.x;
  const int tid = threadIdx.x;
  const int wave = tid >> 5;
  const int lane = tid & 31;
  const __bf16* T = totals + (size_t)b * 64 * NCH;

  __shared__ float G[64][64];
  __shared__ float rsum;
  __shared__ float lsum;

  const int mrow = lane & 15;
  const int aO0 = (lane < 16) ? 0 : 16;
  const int aO1 = aO0 + 8;
  const int bO  = (lane < 16) ? 0 : 16;
  const int mBase = (lane < 16) ? 0 : 8;

  // tiles (ti,tj): wave handles idx=wave and idx=wave+8 -> same tj, shared B
  const int tj  = wave & 3;
  const int ti0 = wave >> 2;        // 0 or 1
  const int ti1 = ti0 + 2;          // 2 or 3
  const __bf16* ar0 = T + (size_t)(ti0 * 16 + mrow) * NCH;
  const __bf16* ar1 = T + (size_t)(ti1 * 16 + mrow) * NCH;
  const __bf16* br  = T + (size_t)(tj  * 16 + mrow) * NCH;
  __builtin_prefetch(ar0, 0, 1);
  __builtin_prefetch(ar1, 0, 1);
  __builtin_prefetch(br, 0, 1);
  v8f acc0 = {};
  v8f acc1 = {};
  for (int k0 = 0; k0 < NCH; k0 += 32) {
    v16bf Bm = cat8(*(const v8bf*)(br + k0 + bO),
                    *(const v8bf*)(br + k0 + bO + 8));
    v16bf A0 = cat8(*(const v8bf*)(ar0 + k0 + aO0),
                    *(const v8bf*)(ar0 + k0 + aO1));
    v16bf A1 = cat8(*(const v8bf*)(ar1 + k0 + aO0),
                    *(const v8bf*)(ar1 + k0 + aO1));
    acc0 = __builtin_amdgcn_wmma_f32_16x16x32_bf16(
        false, A0, false, Bm, (short)0, acc0, false, false);
    acc1 = __builtin_amdgcn_wmma_f32_16x16x32_bf16(
        false, A1, false, Bm, (short)0, acc1, false, false);
  }
#pragma unroll
  for (int r = 0; r < 8; ++r) {
    G[ti0 * 16 + r + mBase][tj * 16 + mrow] = acc0[r];
    G[ti1 * 16 + r + mBase][tj * 16 + mrow] = acc1[r];
  }
  __syncthreads();

  // bandwidth: sum(L2) / (n^2 - n) / KERNEL_MUL^(KERNEL_NUM/2)
  float part = 0.0f;
  for (int e = tid; e < 64 * 64; e += 256) {
    const int i = e >> 6, j = e & 63;
    part += G[i][i] + G[j][j] - 2.0f * G[i][j];
  }
  if (tid == 0) { rsum = 0.0f; lsum = 0.0f; }
  __syncthreads();
  atomicAdd(&rsum, part);
  __syncthreads();
  const float bw = rsum / (float)(64 * 64 - 64) / 4.0f;

  float lpart = 0.0f;
  if (tid < NSAMP) {
    const int i = tid, j = (tid + 1) & (NSAMP - 1);
    auto L2 = [&](int x, int y) { return G[x][x] + G[y][y] - 2.0f * G[x][y]; };
    auto kv = [&](float x) {
      float s = 0.0f, bq = bw;
#pragma unroll
      for (int q = 0; q < 5; ++q) { s += expf(-x / bq); bq *= 2.0f; }
      return s;
    };
    lpart = kv(L2(i, j)) + kv(L2(i + NSAMP, j + NSAMP))
          - kv(L2(i, j + NSAMP)) - kv(L2(j, i + NSAMP));
  }
  atomicAdd(&lsum, lpart);
  __syncthreads();
  if (tid == 0) atomicAdd(&accums[2], lsum / (float)NSAMP);
}

// ---------------- kernel D: cross-entropy + combine + emit outputs ----------------
__global__ void dacam_final_kernel(const float* __restrict__ logits,
                                   const int*   __restrict__ target,
                                   const float* __restrict__ accums,
                                   float* __restrict__ out) {
  const int tid = threadIdx.x;
  for (int i = tid; i < NB * NCL; i += 256) out[i] = logits[i];
  __shared__ float ls[NB];
  if (tid < NB) {
    const float* row = logits + (size_t)tid * NCL;
    float mx = -1e30f;
    for (int c = 0; c < NCL; ++c) mx = fmaxf(mx, row[c]);
    float se = 0.0f;
    for (int c = 0; c < NCL; ++c) se += expf(row[c] - mx);
    ls[tid] = (mx + logf(se)) - row[target[tid]];
  }
  __syncthreads();
  if (tid == 0) {
    float lc = 0.0f;
    for (int i = 0; i < NB; ++i) lc += ls[i];
    lc /= (float)NB;
    const float ld = accums[2] / (float)NB;                       // mean over batches
    const float lu = accums[0] / fmaxf(accums[1] * (float)NCH, 1.0f);
    out[NB * NCL] = lc + 0.3f * ld + 2.0f * lu;
  }
}

// ---------------- launcher ----------------
extern "C" void kernel_launch(void* const* d_in, const int* in_sizes, int n_in,
                              void* d_out, int out_size, void* d_ws, size_t ws_size,
                              hipStream_t stream) {
  const float* px     = (const float*)d_in[0];   // [32,2048,28,28]
  const float* imgf   = (const float*)d_in[1];   // [32,2048]
  const float* logits = (const float*)d_in[2];   // [32,200]
  const float* ccent  = (const float*)d_in[3];   // [200,2048]
  const float* ucent  = (const float*)d_in[4];   // [1,2048]
  const int*   target = (const int*)d_in[5];     // [32]

  char* ws = (char*)d_ws;
  __bf16* featsT = (__bf16*)ws;
  float*  s2     = (float*)(ws + FEATS_BYTES);
  __bf16* totals = (__bf16*)(ws + FEATS_BYTES + S2_BYTES);
  float*  accums = (float*)(ws + FEATS_BYTES + S2_BYTES + TOT_BYTES);
  float*  out    = (float*)d_out;

  dacam_transpose_kernel<<<NB * PT, 256, 0, stream>>>(px, featsT, s2, accums);
  dacam_kmeans_kernel<<<NB, 256, 0, stream>>>(featsT, s2, imgf, ccent, ucent,
                                              target, totals, accums);
  dacam_mmd_kernel<<<NB, 256, 0, stream>>>(totals, accums);
  dacam_final_kernel<<<1, 256, 0, stream>>>(logits, target, accums, out);
}
